// DynAAMSCLoss_90580860273216
// MI455X (gfx1250) — matrix operand order repaired and checked by reading, not compiled, not run on hardware
//
#include <hip/hip_runtime.h>
#include <hip/hip_bf16.h>
#include <math.h>

typedef __attribute__((ext_vector_type(16))) __bf16 v16bf;
typedef __attribute__((ext_vector_type(8)))  float  v8f;

#define ACC_CE    0
#define ACC_INTRA 1
#define ACC_DIAG  2
#define ACC_MARG  3
#define ACC_INTER 4
#define N_ACC     8

__device__ __forceinline__ float clamp1(float v) {
    return fminf(fmaxf(v, -1.0f), 1.0f);
}

// ---------------------------------------------------------------- init
__global__ void init_acc_kernel(float* acc) {
    if (threadIdx.x < N_ACC) acc[threadIdx.x] = 0.0f;
}

// ---------------------------------------------------------------- CE / intra / diag (per-row)
// One block per row. Streaming online logsumexp over C elements (logits read
// exactly once -> this kernel is the HBM roofline term, ~164MB total).
__global__ void row_reduce_kernel(const float* __restrict__ logits,
                                  const int* __restrict__ label,
                                  const float* __restrict__ W,
                                  float* __restrict__ acc,
                                  int C, int D) {
    const int row = blockIdx.x;
    const int tid = threadIdx.x;
    const float* lr = logits + (size_t)row * C;
    const int y = label[row];

    // online (max, sum-exp)
    float m = -3.402823466e+38f, s = 0.0f;
    for (int j = tid; j < C; j += blockDim.x) {
        float x = lr[j];
        if (x > m) { s = s * __expf(m - x) + 1.0f; m = x; }
        else       { s += __expf(x - m); }
    }

    __shared__ float Ms[256], Ss[256], Ws[256];
    Ms[tid] = m; Ss[tid] = s;
    __syncthreads();
    for (int off = 128; off > 0; off >>= 1) {
        if (tid < off) {
            float m1 = Ms[tid], s1 = Ss[tid];
            float m2 = Ms[tid + off], s2 = Ss[tid + off];
            float M = fmaxf(m1, m2);
            Ms[tid] = M;
            Ss[tid] = s1 * __expf(m1 - M) + s2 * __expf(m2 - M);
        }
        __syncthreads();
    }

    // |w_y|^2 for the diagonal (j == y_i) exclusion term, in f32
    float wv = 0.0f;
    for (int t = tid; t < D; t += blockDim.x) {
        float w = W[(size_t)y * D + t];
        wv += w * w;
    }
    Ws[tid] = wv;
    __syncthreads();
    for (int off = 128; off > 0; off >>= 1) {
        if (tid < off) Ws[tid] += Ws[tid + off];
        __syncthreads();
    }

    if (tid == 0) {
        float lse = Ms[0] + logf(Ss[0]);
        float z = lr[y];
        atomicAdd(&acc[ACC_CE], lse - z);                     // -log p[y]
        atomicAdd(&acc[ACC_INTRA], acosf(clamp1(z * 10.0f))); // z / 0.1
        atomicAdd(&acc[ACC_DIAG], acosf(clamp1(Ws[0])));      // A[i, y_i]
    }
}

// ---------------------------------------------------------------- margins mean
__global__ void margins_kernel(const float* __restrict__ margins, float* __restrict__ acc, int C) {
    __shared__ float red[256];
    float s = 0.0f;
    for (int j = threadIdx.x; j < C; j += blockDim.x) s += margins[j];
    red[threadIdx.x] = s;
    __syncthreads();
    for (int off = 128; off > 0; off >>= 1) {
        if (threadIdx.x < off) red[threadIdx.x] += red[threadIdx.x + off];
        __syncthreads();
    }
    if (threadIdx.x == 0) atomicAdd(&acc[ACC_MARG], red[0]);
}

// ---------------------------------------------------------------- inter-class GEMM (WMMA)
// S = W[label] @ W^T  (M=B x K=D) x (K=D x N=C), fused acos(clip())+sum epilogue.
// Specialized path: D is a compile-time constant (fully unrolled K) and each
// wave computes an NT-wide strip of 16x16 tiles (16 x 16*NT output), reusing
// its A fragment NT times per K-step -> NT * D/32 straight-line v_wmma ops.
// 8 waves / block. Fragments hold contiguous K (k = 16*khalf + e) in both A and B.
template<int D, int NT>
__global__ void inter_gemm_strip_kernel(const float* __restrict__ W,
                                        const int* __restrict__ label,
                                        float* __restrict__ acc) {
    const int lane  = threadIdx.x & 31;
    const int wave  = threadIdx.x >> 5;
    const int nbase = blockIdx.x * (NT * 16);          // first column tile (class j)
    const int m0    = (blockIdx.y * 8 + wave) * 16;    // row tile (sample i)

    const int mrow  = lane & 15;   // M (for A) / N (for B) within tile
    const int khalf = lane >> 4;   // which K-half this lane holds

    const int ya = label[m0 + mrow];
    const float* __restrict__ arow = W + (size_t)ya * D;   // A[m,:] = W[y_m,:]

    const float* brow[NT];
    #pragma unroll
    for (int t = 0; t < NT; ++t)
        brow[t] = W + (size_t)(nbase + t * 16 + mrow) * D; // B[:,n] = W[n,:]

    v8f c[NT];
    #pragma unroll
    for (int t = 0; t < NT; ++t) c[t] = (v8f){};

    #pragma unroll
    for (int kb = 0; kb < D; kb += 32) {
        // A: e in [0,16): K = kb + 16*khalf + e (contiguous 16)
        v16bf a;
        const float* ap0 = arow + kb + 16 * khalf;
        const float* ap1 = arow + kb + 16 * khalf + 8;
        #pragma unroll
        for (int e = 0; e < 8; ++e) {
            a[e]     = (__bf16)ap0[e];
            a[e + 8] = (__bf16)ap1[e];
        }
        // one A fragment feeds NT WMMAs
        #pragma unroll
        for (int t = 0; t < NT; ++t) {
            v16bf b;   // element e holds K = kb + 16*khalf + e (contiguous 16)
            const float* bp = brow[t] + kb + 16 * khalf;
            #pragma unroll
            for (int e = 0; e < 16; ++e) b[e] = (__bf16)bp[e];
            c[t] = __builtin_amdgcn_wmma_f32_16x16x32_bf16(
                       /*neg_a=*/false, a, /*neg_b=*/false, b,
                       /*c_mod=*/(short)0, c[t], /*reuse_a=*/false, /*reuse_b=*/false);
        }
    }

    // epilogue: sum acos(clip(S)) over the strip
    float s = 0.0f;
    #pragma unroll
    for (int t = 0; t < NT; ++t) {
        #pragma unroll
        for (int e = 0; e < 8; ++e) s += acosf(clamp1(c[t][e]));
    }

    __shared__ float red[256];
    red[threadIdx.x] = s;
    __syncthreads();
    for (int off = 128; off > 0; off >>= 1) {
        if (threadIdx.x < off) red[threadIdx.x] += red[threadIdx.x + off];
        __syncthreads();
    }
    if (threadIdx.x == 0) atomicAdd(&acc[ACC_INTER], red[0]);
}

// Generic fallback: one 16x16 tile per wave, runtime D (multiple of 32).
__global__ void inter_gemm_kernel(const float* __restrict__ W,
                                  const int* __restrict__ label,
                                  float* __restrict__ acc,
                                  int C, int D) {
    const int lane  = threadIdx.x & 31;
    const int wave  = threadIdx.x >> 5;
    const int n0    = blockIdx.x * 16;
    const int m0    = (blockIdx.y * 8 + wave) * 16;

    const int mrow  = lane & 15;
    const int khalf = lane >> 4;

    const int ya = label[m0 + mrow];
    const float* __restrict__ arow = W + (size_t)ya * D;
    const float* __restrict__ brow = W + (size_t)(n0 + mrow) * D;

    v8f c = {};
    for (int kb = 0; kb < D; kb += 32) {
        v16bf a, b;
        const float* ap0 = arow + kb + 16 * khalf;
        const float* ap1 = arow + kb + 16 * khalf + 8;
        #pragma unroll
        for (int e = 0; e < 8; ++e) {
            a[e]     = (__bf16)ap0[e];
            a[e + 8] = (__bf16)ap1[e];
        }
        const float* bp = brow + kb + 16 * khalf;
        #pragma unroll
        for (int e = 0; e < 16; ++e) b[e] = (__bf16)bp[e];
        c = __builtin_amdgcn_wmma_f32_16x16x32_bf16(
                false, a, false, b, (short)0, c, false, false);
    }

    float s = 0.0f;
    #pragma unroll
    for (int e = 0; e < 8; ++e) s += acosf(clamp1(c[e]));

    __shared__ float red[256];
    red[threadIdx.x] = s;
    __syncthreads();
    for (int off = 128; off > 0; off >>= 1) {
        if (threadIdx.x < off) red[threadIdx.x] += red[threadIdx.x + off];
        __syncthreads();
    }
    if (threadIdx.x == 0) atomicAdd(&acc[ACC_INTER], red[0]);
}

// ---------------------------------------------------------------- finalize
__global__ void finalize_kernel(const float* __restrict__ acc, float* __restrict__ out,
                                int B, int C) {
    const float PI = 3.14159265358979323846f;
    float ce     = acc[ACC_CE] / (float)B;
    float margin = 0.1f * acc[ACC_MARG] / (float)C;
    float intra  = acc[ACC_INTRA] / ((float)B * PI);
    float inter  = (acc[ACC_INTER] - acc[ACC_DIAG]) /
                   ((float)B * (float)(C - 1) * PI);
    out[0] = ce + margin + intra + inter;
}

// ---------------------------------------------------------------- launch
extern "C" void kernel_launch(void* const* d_in, const int* in_sizes, int n_in,
                              void* d_out, int out_size, void* d_ws, size_t ws_size,
                              hipStream_t stream) {
    const float* logits  = (const float*)d_in[0];
    const float* margins = (const float*)d_in[1];
    const float* weights = (const float*)d_in[2];
    const int*   label   = (const int*)d_in[3];

    const int B = in_sizes[3];            // 4096
    const int C = in_sizes[1];            // 10000
    const int D = in_sizes[2] / C;        // 128

    float* acc = (float*)d_ws;
    float* out = (float*)d_out;

    init_acc_kernel<<<1, 32, 0, stream>>>(acc);

    row_reduce_kernel<<<B, 256, 0, stream>>>(logits, label, weights, acc, C, D);

    margins_kernel<<<1, 256, 0, stream>>>(margins, acc, C);

    constexpr int NT = 5;                 // 10000 = 125 * (5*16)
    if (D == 128 && (C % (NT * 16)) == 0 && (B % 128) == 0) {
        dim3 ggrid(C / (NT * 16), B / 128);
        inter_gemm_strip_kernel<128, NT><<<ggrid, 256, 0, stream>>>(weights, label, acc);
    } else {
        dim3 ggrid(C / 16, B / 128);
        inter_gemm_kernel<<<ggrid, 256, 0, stream>>>(weights, label, acc, C, D);
    }

    finalize_kernel<<<1, 1, 0, stream>>>(acc, out, B, C);
}
